// GINClassifier_50397146251361
// MI455X (gfx1250) — matrix operand order repaired and checked by reading, not compiled, not run on hardware
//
#include <hip/hip_runtime.h>

#define NNODES   100000
#define NEDGES   1600000
#define INDIM    128
#define HID      256
#define NLAYERS  3
#define OUTDIM   10
#define NGRAPHS  512
#define BN_EPS   1e-5f
#define CATDIM   (HID * NLAYERS)

typedef __bf16 bf16_t;
typedef __attribute__((ext_vector_type(16))) __bf16 v16bf;
typedef __attribute__((ext_vector_type(8)))  float  v8f;

// ---------------------------------------------------------------------------
// One-time weight prep: f32 [K][N] row-major  ->  bf16 [N][K] (WMMA-B friendly)
// ---------------------------------------------------------------------------
__global__ void transpose_bf16_kernel(const float* __restrict__ W,
                                      bf16_t* __restrict__ Wt, int K, int N) {
  int tid = blockIdx.x * blockDim.x + threadIdx.x;
  if (tid >= K * N) return;
  int n = tid / K;
  int k = tid - n * K;
  Wt[tid] = (bf16_t)W[(size_t)k * N + n];
}

// ---------------------------------------------------------------------------
// A-fragment loader: 16x32 bf16 tile row m0+lane%16, K window [k0, k0+32).
// lane<16 holds K in [0,16); lane>=16 holds [16,32) (matches B's contiguous order).
// `arow` is pre-offset by (lane>>4)*16 within the row.
// ---------------------------------------------------------------------------
__device__ __forceinline__ v16bf load_a_frag(const float* __restrict__ ap) {
  float4 c0 = *(const float4*)(ap);
  float4 c1 = *(const float4*)(ap + 4);
  float4 c2 = *(const float4*)(ap + 8);
  float4 c3 = *(const float4*)(ap + 12);
  v16bf af;
  af[0]  = (bf16_t)c0.x; af[1]  = (bf16_t)c0.y; af[2]  = (bf16_t)c0.z; af[3]  = (bf16_t)c0.w;
  af[4]  = (bf16_t)c1.x; af[5]  = (bf16_t)c1.y; af[6]  = (bf16_t)c1.z; af[7]  = (bf16_t)c1.w;
  af[8]  = (bf16_t)c2.x; af[9]  = (bf16_t)c2.y; af[10] = (bf16_t)c2.z; af[11] = (bf16_t)c2.w;
  af[12] = (bf16_t)c3.x; af[13] = (bf16_t)c3.y; af[14] = (bf16_t)c3.z; af[15] = (bf16_t)c3.w;
  return af;
}

// ---------------------------------------------------------------------------
// WMMA GEMM: C[M x 256] = relu( BN?( A[M x K] @ B + bias ) )
//   A: f32 row-major.  Bt: bf16 [256][K] (pre-transposed).
//   Block = 128 threads (4 waves). Block covers 32 rows x 256 cols;
//   wave w covers cols [64w, 64w+64) over both 16-row stripes (B reused 2x).
//   M % 32 == 0, K % 32 == 0.
// ---------------------------------------------------------------------------
template <int K>
__global__ void __launch_bounds__(128)
wmma_gemm256_kernel(const float* __restrict__ A, const bf16_t* __restrict__ Bt,
                    float* __restrict__ C, const float* __restrict__ bias,
                    const float* __restrict__ gamma, const float* __restrict__ beta,
                    const float* __restrict__ mean, const float* __restrict__ var) {
  const int lane = threadIdx.x & 31;
  const int wave = threadIdx.x >> 5;
  const int m0   = blockIdx.x << 5;     // 32-row block tile
  const int n0   = wave << 6;           // 64-col wave slab
  const int l15  = lane & 15;
  const int hi   = lane >> 4;           // 0 for lanes 0-15, 1 for lanes 16-31

  v8f acc[2][4] = {};

  const float*  arow0 = A  + (size_t)(m0 + l15) * K + (hi << 4);
  const float*  arow1 = arow0 + (size_t)16 * K;
  // B-fragment base: column = n0 + lane%16 ; 16 contiguous K per lane
  const bf16_t* bcol  = Bt + (size_t)(n0 + l15) * K + (hi << 4);

#pragma unroll
  for (int k0 = 0; k0 < K; k0 += 32) {
    v16bf af0 = load_a_frag(arow0 + k0);
    v16bf af1 = load_a_frag(arow1 + k0);

    const bf16_t* bp = bcol + k0;
    v16bf b0 = *(const v16bf*)(bp);
    v16bf b1 = *(const v16bf*)(bp + (size_t)16 * K);
    v16bf b2 = *(const v16bf*)(bp + (size_t)32 * K);
    v16bf b3 = *(const v16bf*)(bp + (size_t)48 * K);

    acc[0][0] = __builtin_amdgcn_wmma_f32_16x16x32_bf16(false, af0, false, b0, (short)0, acc[0][0], false, false);
    acc[0][1] = __builtin_amdgcn_wmma_f32_16x16x32_bf16(false, af0, false, b1, (short)0, acc[0][1], false, false);
    acc[0][2] = __builtin_amdgcn_wmma_f32_16x16x32_bf16(false, af0, false, b2, (short)0, acc[0][2], false, false);
    acc[0][3] = __builtin_amdgcn_wmma_f32_16x16x32_bf16(false, af0, false, b3, (short)0, acc[0][3], false, false);
    acc[1][0] = __builtin_amdgcn_wmma_f32_16x16x32_bf16(false, af1, false, b0, (short)0, acc[1][0], false, false);
    acc[1][1] = __builtin_amdgcn_wmma_f32_16x16x32_bf16(false, af1, false, b1, (short)0, acc[1][1], false, false);
    acc[1][2] = __builtin_amdgcn_wmma_f32_16x16x32_bf16(false, af1, false, b2, (short)0, acc[1][2], false, false);
    acc[1][3] = __builtin_amdgcn_wmma_f32_16x16x32_bf16(false, af1, false, b3, (short)0, acc[1][3], false, false);
  }

  // Epilogue: C/D layout -> VGPR r, lanes 0-15: row +r ; lanes 16-31: row +r+8
#pragma unroll
  for (int t = 0; t < 4; ++t) {
    int col = n0 + (t << 4) + l15;
    float b = bias ? bias[col] : 0.0f;
    float s = 1.0f, o = b;
    if (gamma) {   // fold BatchNorm (eval) into scale/shift
      float sc = gamma[col] * rsqrtf(var[col] + BN_EPS);
      o = beta[col] + (b - mean[col]) * sc;
      s = sc;
    }
#pragma unroll
    for (int u = 0; u < 2; ++u) {
      const int rbase = m0 + (u << 4) + (hi << 3);
#pragma unroll
      for (int r = 0; r < 8; ++r) {
        float v = fmaf(acc[u][t][r], s, o);
        v = fmaxf(v, 0.0f);
        C[(size_t)(rbase + r) * HID + col] = v;
      }
    }
  }
}

// ---------------------------------------------------------------------------
// Elementwise helpers
// ---------------------------------------------------------------------------
__global__ void copy4_kernel(const float4* __restrict__ s, float4* __restrict__ d, int n) {
  int i = blockIdx.x * blockDim.x + threadIdx.x;
  if (i < n) d[i] = s[i];
}

__global__ void zero4_kernel(float4* __restrict__ d, int n) {
  int i = blockIdx.x * blockDim.x + threadIdx.x;
  if (i < n) d[i] = make_float4(0.f, 0.f, 0.f, 0.f);
}

// ---------------------------------------------------------------------------
// GIN aggregation scatter: agg[dst] += h[src]   (one wave per edge, 8 ch/lane)
// ---------------------------------------------------------------------------
__global__ void edge_scatter_kernel(const float* __restrict__ h, const int* __restrict__ src,
                                    const int* __restrict__ dst, float* __restrict__ agg) {
  int e    = (blockIdx.x * blockDim.x + threadIdx.x) >> 5;
  int lane = threadIdx.x & 31;
  if (e >= NEDGES) return;
  int s = src[e], d = dst[e];
  const float4* hp = (const float4*)(h + (size_t)s * HID + lane * 8);
  float4 v0 = hp[0], v1 = hp[1];
  float* op = agg + (size_t)d * HID + lane * 8;
  atomicAdd(op + 0, v0.x); atomicAdd(op + 1, v0.y);
  atomicAdd(op + 2, v0.z); atomicAdd(op + 3, v0.w);
  atomicAdd(op + 4, v1.x); atomicAdd(op + 5, v1.y);
  atomicAdd(op + 6, v1.z); atomicAdd(op + 7, v1.w);
}

// ---------------------------------------------------------------------------
// global_add_pool into pooled[g][colOff .. colOff+255]  (one wave per node)
// ---------------------------------------------------------------------------
__global__ void pool_kernel(const float* __restrict__ h, const int* __restrict__ batch,
                            float* __restrict__ pooled, int colOff) {
  int node = (blockIdx.x * blockDim.x + threadIdx.x) >> 5;
  int lane = threadIdx.x & 31;
  if (node >= NNODES) return;
  int g = batch[node];
  const float4* hp = (const float4*)(h + (size_t)node * HID + lane * 8);
  float4 v0 = hp[0], v1 = hp[1];
  float* op = pooled + (size_t)g * CATDIM + colOff + lane * 8;
  atomicAdd(op + 0, v0.x); atomicAdd(op + 1, v0.y);
  atomicAdd(op + 2, v0.z); atomicAdd(op + 3, v0.w);
  atomicAdd(op + 4, v1.x); atomicAdd(op + 5, v1.y);
  atomicAdd(op + 6, v1.z); atomicAdd(op + 7, v1.w);
}

// ---------------------------------------------------------------------------
// Tiny FC head (512 rows): one thread per output element, f32
// ---------------------------------------------------------------------------
__global__ void fc_kernel(const float* __restrict__ X, const float* __restrict__ W,
                          const float* __restrict__ b, float* __restrict__ Y,
                          int M, int K, int N, int relu) {
  int tid = blockIdx.x * blockDim.x + threadIdx.x;
  if (tid >= M * N) return;
  int m = tid / N, n = tid - m * N;
  float acc = b[n];
  const float* xr = X + (size_t)m * K;
  for (int k = 0; k < K; ++k) acc = fmaf(xr[k], W[(size_t)k * N + n], acc);
  if (relu) acc = fmaxf(acc, 0.0f);
  Y[tid] = acc;
}

// ---------------------------------------------------------------------------
extern "C" void kernel_launch(void* const* d_in, const int* in_sizes, int n_in,
                              void* d_out, int out_size, void* d_ws, size_t ws_size,
                              hipStream_t stream) {
  const float* x      = (const float*)d_in[0];
  const int*   eidx   = (const int*)  d_in[1];
  const int*   batch  = (const int*)  d_in[2];
  const float* W_emb  = (const float*)d_in[3];
  const float* b_emb  = (const float*)d_in[4];
  const float* Wl1    = (const float*)d_in[5];
  const float* bl1    = (const float*)d_in[6];
  const float* gamma  = (const float*)d_in[7];
  const float* beta   = (const float*)d_in[8];
  const float* rmean  = (const float*)d_in[9];
  const float* rvar   = (const float*)d_in[10];
  const float* Wl2    = (const float*)d_in[11];
  const float* bl2    = (const float*)d_in[12];
  const float* W_fc1  = (const float*)d_in[13];
  const float* b_fc1  = (const float*)d_in[14];
  const float* W_fc2  = (const float*)d_in[15];
  const float* b_fc2  = (const float*)d_in[16];
  const float* W_fc   = (const float*)d_in[17];
  const float* b_fc   = (const float*)d_in[18];

  const int* src = eidx;            // edge_index[0]
  const int* dst = eidx + NEDGES;   // edge_index[1]

  // d_out = (c[512x10], h2[512x128], h[100000x256]) concatenated
  float* out_c  = (float*)d_out;
  float* out_h2 = out_c + NGRAPHS * OUTDIM;
  float* out_h  = out_h2 + NGRAPHS * (HID / 2);

  // workspace layout (all offsets 256B-aligned)
  char*   ws     = (char*)d_ws;
  float*  ws_h   = (float*)ws;                                     // 102,400,000 B
  size_t  off    = (size_t)NNODES * HID * sizeof(float);
  bf16_t* wtEmb  = (bf16_t*)(ws + off);  off += (size_t)HID * INDIM * 2;          // 65,536
  bf16_t* wtL1   = (bf16_t*)(ws + off);  off += (size_t)NLAYERS * HID * HID * 2;  // 393,216
  bf16_t* wtL2   = (bf16_t*)(ws + off);  off += (size_t)NLAYERS * HID * HID * 2;  // 393,216
  float*  pooled = (float*)(ws + off);   off += (size_t)NGRAPHS * CATDIM * 4;     // 1,572,864
  float*  hf     = (float*)(ws + off);                                            // 524,288

  // ---- weight prep: transpose + bf16 convert --------------------------------
  transpose_bf16_kernel<<<(INDIM * HID + 255) / 256, 256, 0, stream>>>(W_emb, wtEmb, INDIM, HID);
  for (int l = 0; l < NLAYERS; ++l) {
    transpose_bf16_kernel<<<(HID * HID + 255) / 256, 256, 0, stream>>>(
        Wl1 + (size_t)l * HID * HID, wtL1 + (size_t)l * HID * HID, HID, HID);
    transpose_bf16_kernel<<<(HID * HID + 255) / 256, 256, 0, stream>>>(
        Wl2 + (size_t)l * HID * HID, wtL2 + (size_t)l * HID * HID, HID, HID);
  }

  // ---- node embedding: h0 = relu(x @ W_emb + b_emb) -> ws_h ----------------
  wmma_gemm256_kernel<INDIM><<<NNODES / 32, 128, 0, stream>>>(
      x, wtEmb, ws_h, b_emb, nullptr, nullptr, nullptr, nullptr);

  // ---- zero pooled accumulator ---------------------------------------------
  zero4_kernel<<<(NGRAPHS * CATDIM / 4 + 255) / 256, 256, 0, stream>>>(
      (float4*)pooled, NGRAPHS * CATDIM / 4);

  // ---- 3 GIN layers; ping-pong ws_h <-> out_h so final h lands in d_out ----
  float* h = ws_h;
  for (int l = 0; l < NLAYERS; ++l) {
    float* agg = (h == ws_h) ? out_h : ws_h;
    // agg = h (eps = 0 -> (1+eps)*h == h)
    copy4_kernel<<<(NNODES * HID / 4 + 255) / 256, 256, 0, stream>>>(
        (const float4*)h, (float4*)agg, NNODES * HID / 4);
    // agg += segment_sum(h[src], dst)
    edge_scatter_kernel<<<(NEDGES * 32 + 255) / 256, 256, 0, stream>>>(h, src, dst, agg);
    // z = relu(BN(agg @ Wl1[l] + bl1[l]))  (h's buffer is dead -> reuse)
    float* z = h;
    wmma_gemm256_kernel<HID><<<NNODES / 32, 128, 0, stream>>>(
        agg, wtL1 + (size_t)l * HID * HID, z, bl1 + l * HID,
        gamma + l * HID, beta + l * HID, rmean + l * HID, rvar + l * HID);
    // h' = relu(z @ Wl2[l] + bl2[l])  (agg's buffer is dead -> reuse)
    float* hn = agg;
    wmma_gemm256_kernel<HID><<<NNODES / 32, 128, 0, stream>>>(
        z, wtL2 + (size_t)l * HID * HID, hn, bl2 + l * HID,
        nullptr, nullptr, nullptr, nullptr);
    // pooled[:, l*256 : (l+1)*256] += segment_sum(h', batch)
    pool_kernel<<<(NNODES * 32 + 255) / 256, 256, 0, stream>>>(hn, batch, pooled, l * HID);
    h = hn;
  }
  // parity: emb->ws_h, L0 h->out_h, L1 h->ws_h, L2 h->out_h  (== d_out slot)

  // ---- FC head --------------------------------------------------------------
  fc_kernel<<<(NGRAPHS * HID + 255) / 256, 256, 0, stream>>>(
      pooled, W_fc1, b_fc1, hf, NGRAPHS, CATDIM, HID, 1);
  fc_kernel<<<(NGRAPHS * (HID / 2) + 255) / 256, 256, 0, stream>>>(
      hf, W_fc2, b_fc2, out_h2, NGRAPHS, HID, HID / 2, 1);
  fc_kernel<<<(NGRAPHS * OUTDIM + 255) / 256, 256, 0, stream>>>(
      out_h2, W_fc, b_fc, out_c, NGRAPHS, HID / 2, OUTDIM, 0);
}
